// TsRnn_34471407517909
// MI455X (gfx1250) — matrix operand order repaired and checked by reading, not compiled, not run on hardware
//
#include <hip/hip_runtime.h>

// Two-layer LSTM rollout, persistent per-batch-chunk workgroups.
// Weights resident in LDS; recurrence GEMM via v_wmma_f32_16x16x32_bf16.

#define TT    512
#define BB    4096
#define CELL  100
#define NG    400      // 4*CELL gates
#define FUT   64
#define WSTR  136      // W row stride in bf16 elems (K padded 128 + 8 anti-bank-conflict)
#define ASTR  136      // h1 (A) row stride in bf16 elems
#define CSTR  104      // c1 row stride in floats
#define GTSTR 20       // transposed gates row stride in floats (16 + 4 pad)
#define MW    16       // batch rows per workgroup
#define NTILES 25      // 400/16
#define THREADS 256    // 8 waves (wave32)

typedef __attribute__((ext_vector_type(16))) __bf16 v16bf;
typedef __attribute__((ext_vector_type(8)))  __bf16 v8bf;
typedef __attribute__((ext_vector_type(8)))  float  v8f;
typedef __attribute__((ext_vector_type(4)))  float  v4f;

// Fast activations: v_exp_f32 + v_rcp_f32 (both TRANS ops, co-execute with VALU).
__device__ __forceinline__ float fsig(float x)  {
    return __builtin_amdgcn_rcpf(1.0f + __expf(-x));
}
__device__ __forceinline__ float ftanh(float x) {
    return 1.0f - 2.0f * __builtin_amdgcn_rcpf(__expf(2.0f * x) + 1.0f);
}

__global__ __launch_bounds__(THREADS)
void TsRnn_lstm_kernel(const float* __restrict__ input,
                       const float* __restrict__ W_ih1,
                       const float* __restrict__ W_hh1,
                       const float* __restrict__ b_ih1,
                       const float* __restrict__ b_hh1,
                       const float* __restrict__ W_ih2,
                       const float* __restrict__ W_hh2,
                       const float* __restrict__ b_ih2,
                       const float* __restrict__ b_hh2,
                       float* __restrict__ out)
{
    // ~157 KB LDS per workgroup (CDNA5: 320 KB per WGP -> 2 WGs resident).
    __shared__ __align__(16) __bf16 Wlds[NG * WSTR];     // W_hh1 as [N=400][Kpad] bf16
    __shared__ __align__(16) __bf16 Alds[MW * ASTR];     // h1 bf16 staging [16][Kpad]
    __shared__ __align__(16) float  c1g[MW * CSTR];      // c1 fp32 master
    __shared__ __align__(16) float  gatesT[NG * GTSTR];  // gate pre-activations, [N][M] transposed
    __shared__ __align__(16) float  g0[NG];              // b_ih1 + b_hh1
    __shared__ __align__(16) float  wx[NG];              // W_ih1 (input size 1)
    __shared__ __align__(16) float  W2[4 * CSTR];        // W_ih2 [4][100]
    __shared__ __align__(32) float  xbuf[MW];            // per-step input x
    __shared__ float g2[MW * 4];                         // lstm2 gate accumulators
    __shared__ float h2s[MW], c2s[MW], b2[4], wh2[4];

    const int tid   = threadIdx.x;
    const int lane  = tid & 31;
    const int wave  = tid >> 5;
    const int half  = lane >> 4;   // 0: lanes 0-15, 1: lanes 16-31
    const int lp    = lane & 15;
    const int bbase = blockIdx.x * MW;

    // ---------------- one-time init: stage weights into LDS ----------------
    for (int i = tid; i < NG * WSTR; i += THREADS) {
        int n = i / WSTR, k = i - n * WSTR;
        Wlds[i] = (k < CELL) ? (__bf16)W_hh1[n * CELL + k] : (__bf16)0.0f;
    }
    for (int i = tid; i < MW * ASTR; i += THREADS) Alds[i] = (__bf16)0.0f; // h1 = 0 (incl K pad)
    for (int i = tid; i < MW * CSTR; i += THREADS) c1g[i] = 0.0f;
    for (int i = tid; i < NG; i += THREADS) { g0[i] = b_ih1[i] + b_hh1[i]; wx[i] = W_ih1[i]; }
    for (int i = tid; i < 4 * CSTR; i += THREADS) {
        int j = i / CSTR, k = i - j * CSTR;
        W2[i] = (k < CELL) ? W_ih2[j * CELL + k] : 0.0f;
    }
    if (tid < 4)  { b2[tid] = b_ih2[tid] + b_hh2[tid]; wh2[tid] = W_hh2[tid]; }
    if (tid < MW) { h2s[tid] = 0.0f; c2s[tid] = 0.0f; }
    __syncthreads();

    // ---------------- 576 sequential timesteps ----------------
    for (int step = 0; step < TT + FUT; ++step) {
        // stage 1: x for this step + init lstm2 gate accumulators
        if (tid < MW) {
            xbuf[tid] = (step < TT) ? input[step * BB + bbase + tid] : c2s[tid];
        } else if (tid >= 32 && tid < 32 + MW * 4) {
            int i = tid - 32, b = i >> 2, j = i & 3;
            g2[i] = b2[j] + h2s[b] * wh2[j];
        } else if (tid >= 96 && tid < 96 + MW && step + 1 < TT) {
            __builtin_prefetch(&input[(step + 1) * BB + bbase + (tid - 96)], 0, 0);
        }
        __syncthreads();

        // stage 2: gates[16,400] = h1[16,128] @ Whh[128,400] + bias + x*W_ih1
        // A frag (16-bit 16x32, ISA 7.12.2): lane M=lp; elems 0-7 -> K = half*8+e,
        // elems 8-15 -> K = 16+half*8+(e-8).
        v16bf afrag[4];
#pragma unroll
        for (int kk = 0; kk < 4; ++kk) {
            const __bf16* ap = &Alds[lp * ASTR + kk * 32 + half * 8];
            v8bf lo = *(const v8bf*)ap;
            v8bf hi = *(const v8bf*)(ap + 16);
            v16bf a;
#pragma unroll
            for (int e = 0; e < 8; ++e) { a[e] = lo[e]; a[8 + e] = hi[e]; }
            afrag[kk] = a;
        }
        v8f xv = *(const v8f*)&xbuf[half * 8];   // x for rows M = half*8 .. +7

        for (int nt = wave; nt < NTILES; nt += 8) {   // wave-uniform -> EXEC all-ones
            int n = nt * 16 + lp;                     // gate column (B/C/D: N = lane%16)
            float g0v = g0[n], wxv = wx[n];
            v8f c;
#pragma unroll
            for (int v = 0; v < 8; ++v) c[v] = g0v + xv[v] * wxv; // C rows M = v + 8*half
#pragma unroll
            for (int kk = 0; kk < 4; ++kk) {
                // B frag (32x16, 16-bit): lane N=lp; elems 0-7 -> K = kk*32 + half*8 + e, elems 8-15 -> K = kk*32 + 16 + half*8 + (e-8)
                const __bf16* bp = &Wlds[n * WSTR + kk * 32 + half * 8];
                v8bf lo = *(const v8bf*)bp;
                v8bf hi = *(const v8bf*)(bp + 16);
                v16bf bfrag;
#pragma unroll
                for (int e = 0; e < 8; ++e) { bfrag[e] = lo[e]; bfrag[8 + e] = hi[e]; }
                c = __builtin_amdgcn_wmma_f32_16x16x32_bf16(
                        false, afrag[kk], false, bfrag, (short)0, c, false, false);
            }
            // transposed store: lane's 8 floats are M = 8*half .. +7 at fixed N
            // -> 32 contiguous bytes -> 2x ds_store_b128
            float* gp = &gatesT[n * GTSTR + 8 * half];
            v4f s0 = {c[0], c[1], c[2], c[3]};
            v4f s1 = {c[4], c[5], c[6], c[7]};
            *(v4f*)gp       = s0;
            *(v4f*)(gp + 4) = s1;
        }
        __syncthreads();

        // stage 3: LSTM1 elementwise: c1, h1 update (h1 -> bf16 A staging).
        // 16-thread group per batch row: b uniform, no div/mod in the loop.
        {
            int b = tid >> 4, l16 = tid & 15;
            for (int u = l16; u < CELL; u += 16) {
                float ig = gatesT[u * GTSTR + b];
                float fg = gatesT[(CELL + u) * GTSTR + b];
                float gg = gatesT[(2 * CELL + u) * GTSTR + b];
                float og = gatesT[(3 * CELL + u) * GTSTR + b];
                float co = c1g[b * CSTR + u];
                float cn = fsig(fg) * co + fsig(ig) * ftanh(gg);
                float hn = fsig(og) * ftanh(cn);
                c1g[b * CSTR + u] = cn;
                Alds[b * ASTR + u] = (__bf16)hn;
            }
        }
        __syncthreads();

        // stage 4: LSTM2 gates = c1 @ W_ih2.T (256 threads x 25-FMA partials)
        {
            int b = tid >> 4, j = (tid >> 2) & 3, kc = tid & 3;
            int kb = kc * 25;
            float acc = 0.0f;
#pragma unroll
            for (int k = 0; k < 25; ++k)
                acc += c1g[b * CSTR + kb + k] * W2[j * CSTR + kb + k];
            atomicAdd(&g2[b * 4 + j], acc);   // ds_add_f32
        }
        __syncthreads();

        // stage 5: LSTM2 state update (+ emit output in future phase)
        if (tid < MW) {
            int b = tid;
            float i2 = g2[b * 4 + 0], f2 = g2[b * 4 + 1];
            float gg2 = g2[b * 4 + 2], o2 = g2[b * 4 + 3];
            float c2n = fsig(f2) * c2s[b] + fsig(i2) * ftanh(gg2);
            float h2n = fsig(o2) * ftanh(c2n);
            c2s[b] = c2n; h2s[b] = h2n;
            if (step >= TT) out[(bbase + b) * FUT + (step - TT)] = c2n;
        }
        __syncthreads();
    }
}

extern "C" void kernel_launch(void* const* d_in, const int* in_sizes, int n_in,
                              void* d_out, int out_size, void* d_ws, size_t ws_size,
                              hipStream_t stream) {
    const float* input = (const float*)d_in[0];
    const float* W_ih1 = (const float*)d_in[1];
    const float* W_hh1 = (const float*)d_in[2];
    const float* b_ih1 = (const float*)d_in[3];
    const float* b_hh1 = (const float*)d_in[4];
    const float* W_ih2 = (const float*)d_in[5];
    const float* W_hh2 = (const float*)d_in[6];
    const float* b_ih2 = (const float*)d_in[7];
    const float* b_hh2 = (const float*)d_in[8];
    float* out = (float*)d_out;
    (void)in_sizes; (void)n_in; (void)out_size; (void)d_ws; (void)ws_size;

    dim3 grid(BB / MW);   // 256 workgroups, 16 batch rows each
    TsRnn_lstm_kernel<<<grid, THREADS, 0, stream>>>(
        input, W_ih1, W_hh1, b_ih1, b_hh1, W_ih2, W_hh2, b_ih2, b_hh2, out);
}
